// EdgeToTriAttention_1778116461007
// MI455X (gfx1250) — matrix operand order repaired and checked by reading, not compiled, not run on hardware
//
#include <hip/hip_runtime.h>

typedef __attribute__((ext_vector_type(16))) __bf16 v16bf;
typedef __attribute__((ext_vector_type(8)))  __bf16 v8bf;
typedef __attribute__((ext_vector_type(8)))  float  v8f;
typedef __attribute__((ext_vector_type(4)))  unsigned int u32x4;
typedef __attribute__((ext_vector_type(8)))  int i32x8;
typedef __attribute__((ext_vector_type(4)))  int i32x4;

static constexpr int   kN     = 8192;
static constexpr int   kDK    = 128;
static constexpr int   kHID   = 256;
static constexpr int   kSEG   = 2048;
static constexpr int   kSPLIT = 8;      // column split for attention (occupancy)
static constexpr float kEPS   = 1.1920929e-07f;            // float32 eps (RMSNorm)
static constexpr float kQK_SCALE = 0.08838834764831845f;   // 1/sqrt(128)
static constexpr float kCLAMP = -126.0f;                   // exp underflow clamp

// ---------------------------------------------------------------------------
// WMMA helper:  D(16x16 f32) = A(16x32 bf16) * B(32x16 bf16) + C
// ---------------------------------------------------------------------------
static __device__ __forceinline__ v8f wmma_bf16(v16bf a, v16bf b, v8f c) {
  return __builtin_amdgcn_wmma_f32_16x16x32_bf16(
      /*neg_a=*/false, a, /*neg_b=*/false, b,
      /*c_mod=*/(short)0, c, /*reuse_a=*/false, /*reuse_b=*/false);
}

// A-operand (16x32, 16-bit, ISA 7.12.2):
//  lanes 0-15 : M=lane,    elems 0..7 -> K=kBase+0..7,  elems 8..15 -> K=kBase+16..23
//  lanes16-31 : M=lane-16, elems 0..7 -> K=kBase+8..15, elems 8..15 -> K=kBase+24..31
static __device__ __forceinline__ v16bf load_a16(const __bf16* buf, int rowStride,
                                                 int rowBase, int kBase, int lane) {
  const int half = lane >> 4, m = lane & 15;
  const __bf16* p = buf + (rowBase + m) * rowStride + kBase + half * 8;
  v8bf lo = *(const v8bf*)(p);
  v8bf hi = *(const v8bf*)(p + 16);
  v16bf r;
#pragma unroll
  for (int i = 0; i < 8; ++i) { r[i] = lo[i]; r[i + 8] = hi[i]; }
  return r;
}

static __device__ __forceinline__ v16bf load_a16_f32(const float* buf, int rowStride,
                                                     int rowBase, int kBase, int lane) {
  const int half = lane >> 4, m = lane & 15;
  const float* p = buf + (size_t)(rowBase + m) * rowStride + kBase + half * 8;
  v16bf r;
#pragma unroll
  for (int i = 0; i < 8; ++i) { r[i] = (__bf16)p[i]; r[i + 8] = (__bf16)p[i + 16]; }
  return r;
}

// B-operand (32x16 KxN, 16-bit): buffer must be [N][K] row-major.
//  lanes 0-15 : N=lane,    elems 0..7 -> K=kBase+0..7,  elems 8..15 -> K=kBase+16..23
//  lanes16-31 : N=lane-16, elems 0..7 -> K=kBase+8..15, elems 8..15 -> K=kBase+24..31
static __device__ __forceinline__ v16bf load_b16(const __bf16* buf, int rowStride,
                                                 int nBase, int kBase, int lane) {
  const int half = lane >> 4, n = lane & 15;
  const __bf16* p = buf + (nBase + n) * rowStride + kBase + half * 8;
  v8bf lo = *(const v8bf*)(p);
  v8bf hi = *(const v8bf*)(p + 16);
  v16bf r;
#pragma unroll
  for (int i = 0; i < 8; ++i) { r[i] = lo[i]; r[i + 8] = hi[i]; }
  return r;
}

// Interleaved 16-lane butterflies over 8 independent rows.
static __device__ __forceinline__ void redmax16x8(float v[8]) {
#pragma unroll
  for (int off = 1; off < 16; off <<= 1) {
    float t[8];
#pragma unroll
    for (int r = 0; r < 8; ++r) t[r] = __shfl_xor(v[r], off, 32);
#pragma unroll
    for (int r = 0; r < 8; ++r) v[r] = fmaxf(v[r], t[r]);
  }
}
static __device__ __forceinline__ void redsum16x8(float v[8]) {
#pragma unroll
  for (int off = 1; off < 16; off <<= 1) {
    float t[8];
#pragma unroll
    for (int r = 0; r < 8; ++r) t[r] = __shfl_xor(v[r], off, 32);
#pragma unroll
    for (int r = 0; r < 8; ++r) v[r] += t[r];
  }
}

// ---------------------------------------------------------------------------
// Tensor Data Mover: DMA a 2-D bf16 tile (tile_d1 rows x tile_d0 elems) from a
// row-major global tensor (row stride = stride0_elems) into contiguous LDS.
// D# layout per CDNA5 ISA 8.3-8.5 (2-D: groups 2/3 zero).  One instruction per
// wave, tracked by TENSORcnt.
// ---------------------------------------------------------------------------
static __device__ __forceinline__ void tdm_load_bf16_2d(
    unsigned lds_addr, const __bf16* gptr,
    unsigned tile_d0, unsigned tile_d1,
    unsigned long long stride0_elems,
    unsigned long long tensor_d0, unsigned long long tensor_d1) {
  const unsigned long long ga = (unsigned long long)(uintptr_t)gptr;
  u32x4 g0;
  g0[0] = 1u;                                   // count=1 (valid user D#)
  g0[1] = lds_addr;                             // LDS byte address
  g0[2] = (unsigned)(ga & 0xFFFFFFFFu);         // global_addr[31:0]
  g0[3] = (unsigned)((ga >> 32) & 0x01FFFFFFu)  // global_addr[56:32]
          | 0x80000000u;                        // type = 2 ("image")
  i32x8 g1;
  g1[0] = (int)(1u << 16);                                // data_size=1 (2 bytes)
  g1[1] = (int)((tensor_d0 & 0xFFFFu) << 16);             // tensor_dim0[15:0]
  g1[2] = (int)(((tensor_d0 >> 16) & 0xFFFFu) |           // tensor_dim0[31:16]
                ((tensor_d1 & 0xFFFFu) << 16));           // tensor_dim1[15:0]
  g1[3] = (int)(((tensor_d1 >> 16) & 0xFFFFu) |           // tensor_dim1[31:16]
                (tile_d0 << 16));                         // tile_dim0
  g1[4] = (int)tile_d1;                                   // tile_dim1 (tile_dim2=0)
  g1[5] = (int)(stride0_elems & 0xFFFFFFFFu);             // dim0 stride lo32
  g1[6] = (int)((stride0_elems >> 32) & 0xFFFFu);         // dim0 stride hi16
  g1[7] = 0;
  i32x4 gz = {0, 0, 0, 0};
#if defined(__clang_major__) && (__clang_major__ >= 23)
  i32x8 gz8 = {0, 0, 0, 0, 0, 0, 0, 0};
  __builtin_amdgcn_tensor_load_to_lds(g0, g1, gz, gz, gz8, 0);
#else
  __builtin_amdgcn_tensor_load_to_lds(g0, g1, gz, gz, 0);
#endif
}

// ---------------------------------------------------------------------------
// 1) q/k/v projection (K=3 -> pure VALU).  q gets 1/sqrt(dk) folded in.
// ---------------------------------------------------------------------------
__global__ void eta_proj(const float* __restrict__ ef,
                         const float* __restrict__ Wq, const float* __restrict__ bq,
                         const float* __restrict__ Wk, const float* __restrict__ bk,
                         const float* __restrict__ Wv, const float* __restrict__ bv,
                         __bf16* __restrict__ qbf, __bf16* __restrict__ kbf,
                         __bf16* __restrict__ vt) {
  const int idx = blockIdx.x * blockDim.x + threadIdx.x;
  if (idx >= kN * kDK) return;
  const int row = idx >> 7, c = idx & (kDK - 1);
  const float e0 = ef[row * 3 + 0], e1 = ef[row * 3 + 1], e2 = ef[row * 3 + 2];
  const float q = e0 * Wq[c] + e1 * Wq[kDK + c] + e2 * Wq[2 * kDK + c] + bq[c];
  const float k = e0 * Wk[c] + e1 * Wk[kDK + c] + e2 * Wk[2 * kDK + c] + bk[c];
  const float v = e0 * Wv[c] + e1 * Wv[kDK + c] + e2 * Wv[2 * kDK + c] + bv[c];
  qbf[idx] = (__bf16)(q * kQK_SCALE);
  kbf[idx] = (__bf16)k;
  vt[(size_t)c * kN + row] = (__bf16)v;
}

// ---------------------------------------------------------------------------
// 2) Pre-transpose MLP weights to bf16 [out][in].
// ---------------------------------------------------------------------------
__global__ void eta_convw(const float* __restrict__ W0, const float* __restrict__ Wres,
                          __bf16* __restrict__ W0t, __bf16* __restrict__ Wrt) {
  const int idx = blockIdx.x * blockDim.x + threadIdx.x;
  if (idx < kDK * kHID) {
    const int o = idx / kDK, i = idx % kDK;
    W0t[idx] = (__bf16)W0[i * kHID + o];
  }
  const int j = idx - kDK * kHID;
  if (j >= 0 && j < 2 * kHID * kHID) {
    const int res = j / (kHID * kHID);
    const int rem = j % (kHID * kHID);
    const int o = rem / kHID, i = rem % kHID;
    Wrt[j] = (__bf16)Wres[res * kHID * kHID + i * kHID + o];
  }
}

// ---------------------------------------------------------------------------
// 3a) Masked flash-attention, split-K over columns, TDM-staged tiles.
//     grid = (64 row-blocks, kSPLIT column-chunks); block = 256 thr = 8 waves.
// ---------------------------------------------------------------------------
__global__ __launch_bounds__(256) void eta_attn_part(
    const __bf16* __restrict__ qbf, const __bf16* __restrict__ kbf,
    const __bf16* __restrict__ vt, const int* __restrict__ tri,
    float* __restrict__ Opart, float* __restrict__ mpart, float* __restrict__ lpart) {
  __shared__ alignas(16) __bf16 Ks[32 * kDK];   // [col][feat]  (B for S)
  __shared__ alignas(16) __bf16 Vs[kDK * 32];   // [feat][col]  (B for P@V)
  __shared__ alignas(16) __bf16 Ps[8][16 * 32]; // wave-private P re-layout
  __shared__ int tris[32];

  const int t = threadIdx.x;
  const int w = t >> 5, lane = t & 31;
  const int half = lane >> 4, col = lane & 15;
  const int rowBase = blockIdx.x * 128 + w * 16;
  const int cs = blockIdx.y;
  const int colStart = cs * (kN / kSPLIT);
  constexpr int kTiles = (kN / kSPLIT) / 32;

  v16bf aq[4];
#pragma unroll
  for (int kc = 0; kc < 4; ++kc) aq[kc] = load_a16(qbf, kDK, rowBase, kc * 32, lane);
  int rowtri[8];
#pragma unroll
  for (int r = 0; r < 8; ++r) rowtri[r] = tri[rowBase + half * 8 + r];

  const v8f zero = {};
  v8f O[8];
#pragma unroll
  for (int n = 0; n < 8; ++n) O[n] = zero;
  float m[8], l[8];
#pragma unroll
  for (int r = 0; r < 8; ++r) { m[r] = -__builtin_inff(); l[r] = 0.f; }

  for (int jt = 0; jt < kTiles; ++jt) {
    const int cb = colStart + jt * 32;
    __syncthreads();   // everyone done reading previous tile
    if (w == 0) {      // wave 0 drives the Tensor Data Mover (EXEC-independent)
      // K tile: 32 rows x 128 feats, row-major src -> contiguous LDS [col][feat]
      tdm_load_bf16_2d((unsigned)(uintptr_t)Ks, kbf + (size_t)cb * kDK,
                       /*tile_d0=*/kDK, /*tile_d1=*/32,
                       /*stride0=*/kDK, /*tdim0=*/kDK, /*tdim1=*/kN);
      // V tile: 128 feats x 32 cols from vt[feat][col] -> contiguous LDS [feat][col]
      tdm_load_bf16_2d((unsigned)(uintptr_t)Vs, vt + cb,
                       /*tile_d0=*/32, /*tile_d1=*/kDK,
                       /*stride0=*/kN, /*tdim0=*/kN, /*tdim1=*/kDK);
      __builtin_amdgcn_s_wait_tensorcnt(0);
    }
    if (t < 32) tris[t] = tri[cb + t];
    __syncthreads();   // publish Ks/Vs/tris

    // S = Q @ K^T  (two 16x16 tiles, K=128), B loads software-pipelined so the
    // dscnt wait covers a load issued one WMMA earlier.
    v8f s0 = zero, s1 = zero;
    v16bf bk0 = load_b16(Ks, kDK, 0, 0, lane);
    v16bf bk1 = load_b16(Ks, kDK, 16, 0, lane);
#pragma unroll
    for (int kc = 0; kc < 4; ++kc) {
      v16bf nb0 = bk0, nb1 = bk1;
      if (kc < 3) {
        nb0 = load_b16(Ks, kDK, 0,  (kc + 1) * 32, lane);
        nb1 = load_b16(Ks, kDK, 16, (kc + 1) * 32, lane);
      }
      s0 = wmma_bf16(aq[kc], bk0, s0);
      s1 = wmma_bf16(aq[kc], bk1, s1);
      bk0 = nb0; bk1 = nb1;
    }

    // triangle mask + branchless online softmax (interleaved reductions)
    const int ct0 = tris[col], ct1 = tris[16 + col];
    float a0[8], a1[8], tmx[8];
#pragma unroll
    for (int r = 0; r < 8; ++r) {
      a0[r] = (rowtri[r] == ct0) ? s0[r] : -__builtin_inff();
      a1[r] = (rowtri[r] == ct1) ? s1[r] : -__builtin_inff();
      tmx[r] = fmaxf(a0[r], a1[r]);
    }
    redmax16x8(tmx);

    float sc[8], ps[8];
#pragma unroll
    for (int r = 0; r < 8; ++r) {
      const float mn = fmaxf(m[r], tmx[r]);
      sc[r] = __expf(fmaxf(m[r] - mn, kCLAMP));      // NaN-safe, branch-free
      const float p0 = __expf(fmaxf(a0[r] - mn, kCLAMP));
      const float p1 = __expf(fmaxf(a1[r] - mn, kCLAMP));
      ps[r] = p0 + p1;
      m[r] = mn;
      Ps[w][(half * 8 + r) * 32 + col]      = (__bf16)p0;
      Ps[w][(half * 8 + r) * 32 + 16 + col] = (__bf16)p1;
    }
    redsum16x8(ps);
#pragma unroll
    for (int r = 0; r < 8; ++r) {
      l[r] = l[r] * sc[r] + ps[r];
#pragma unroll
      for (int n = 0; n < 8; ++n) O[n][r] *= sc[r];
    }

    // O += P @ V  (pipelined B loads again)
    const v16bf pa = load_a16(Ps[w], 32, 0, 0, lane);
    v16bf bv0 = load_b16(Vs, 32, 0, 0, lane);
#pragma unroll
    for (int n = 0; n < 8; ++n) {
      v16bf nb = bv0;
      if (n < 7) nb = load_b16(Vs, 32, (n + 1) * 16, 0, lane);
      O[n] = wmma_bf16(pa, bv0, O[n]);
      bv0 = nb;
    }
  }

  // write un-normalized partials
#pragma unroll
  for (int n = 0; n < 8; ++n) {
#pragma unroll
    for (int r = 0; r < 8; ++r) {
      const int row = rowBase + half * 8 + r;
      Opart[((size_t)cs * kN + row) * kDK + n * 16 + col] = O[n][r];
    }
  }
  if (col == 0) {
#pragma unroll
    for (int r = 0; r < 8; ++r) {
      const int row = rowBase + half * 8 + r;
      mpart[cs * kN + row] = m[r];
      lpart[cs * kN + row] = l[r];
    }
  }
}

// ---------------------------------------------------------------------------
// 3b) Merge column-split partials.
// ---------------------------------------------------------------------------
__global__ void eta_attn_merge(const float* __restrict__ Opart,
                               const float* __restrict__ mpart,
                               const float* __restrict__ lpart,
                               float* __restrict__ att) {
  const int idx = blockIdx.x * blockDim.x + threadIdx.x;   // row*128 + feat
  if (idx >= kN * kDK) return;
  const int row = idx >> 7;
  float mstar = -__builtin_inff();
#pragma unroll
  for (int cs = 0; cs < kSPLIT; ++cs) mstar = fmaxf(mstar, mpart[cs * kN + row]);
  float acc = 0.f, lsum = 0.f;
#pragma unroll
  for (int cs = 0; cs < kSPLIT; ++cs) {
    const float wgt = __expf(fmaxf(mpart[cs * kN + row] - mstar, kCLAMP));
    acc  += Opart[(size_t)cs * kN * kDK + idx] * wgt;
    lsum += lpart[cs * kN + row] * wgt;
  }
  att[idx] = acc / lsum;
}

// ---------------------------------------------------------------------------
// 4) MLP head, one wave per 16 rows, all GEMMs via WMMA.
// ---------------------------------------------------------------------------
__global__ __launch_bounds__(32) void eta_mlp(
    const float* __restrict__ attended, const __bf16* __restrict__ W0t,
    const float* __restrict__ b0, const float* __restrict__ rms_w,
    const __bf16* __restrict__ Wrt, const float* __restrict__ bres,
    const float* __restrict__ Wout, const float* __restrict__ bout,
    float* __restrict__ logits) {
  __shared__ alignas(16) __bf16 g_lds[16 * kHID];
  const int lane = threadIdx.x & 31;
  const int half = lane >> 4, col = lane & 15;
  const int rowBase = blockIdx.x * 16;

  v16bf a[4];
#pragma unroll
  for (int kc = 0; kc < 4; ++kc) a[kc] = load_a16_f32(attended, kDK, rowBase, kc * 32, lane);

  v8f h[16];
#pragma unroll
  for (int n = 0; n < 16; ++n) {
    v8f acc = {};
#pragma unroll
    for (int kc = 0; kc < 4; ++kc)
      acc = wmma_bf16(a[kc], load_b16(W0t, kDK, n * 16, kc * 32, lane), acc);
    const float bb = b0[n * 16 + col];
#pragma unroll
    for (int r = 0; r < 8; ++r) h[n][r] = acc[r] + bb;
  }

  for (int res = 0; res < 2; ++res) {
    float sq[8];
#pragma unroll
    for (int r = 0; r < 8; ++r) sq[r] = 0.f;
#pragma unroll
    for (int n = 0; n < 16; ++n)
#pragma unroll
      for (int r = 0; r < 8; ++r) sq[r] += h[n][r] * h[n][r];
    redsum16x8(sq);
    float rinv[8];
#pragma unroll
    for (int r = 0; r < 8; ++r) rinv[r] = rsqrtf(sq[r] * (1.f / kHID) + kEPS);

#pragma unroll
    for (int n = 0; n < 16; ++n) {
      const float gw = rms_w[res * kHID + n * 16 + col];
#pragma unroll
      for (int r = 0; r < 8; ++r)
        g_lds[(half * 8 + r) * kHID + n * 16 + col] = (__bf16)(h[n][r] * rinv[r] * gw);
    }
    v16bf ga[8];
#pragma unroll
    for (int kc = 0; kc < 8; ++kc) ga[kc] = load_a16(g_lds, kHID, 0, kc * 32, lane);
    const __bf16* Wr = Wrt + res * kHID * kHID;
#pragma unroll
    for (int n = 0; n < 16; ++n) {
      v8f acc = {};
#pragma unroll
      for (int kc = 0; kc < 8; ++kc)
        acc = wmma_bf16(ga[kc], load_b16(Wr, kHID, n * 16, kc * 32, lane), acc);
      const float bb = bres[res * kHID + n * 16 + col];
#pragma unroll
      for (int r = 0; r < 8; ++r) h[n][r] += fmaxf(acc[r] + bb, 0.f);
    }
  }

  float part[8];
#pragma unroll
  for (int r = 0; r < 8; ++r) part[r] = 0.f;
#pragma unroll
  for (int n = 0; n < 16; ++n) {
    const float wv = Wout[n * 16 + col];
#pragma unroll
    for (int r = 0; r < 8; ++r) part[r] += h[n][r] * wv;
  }
  redsum16x8(part);
  const float bo = bout[0];
  if (col == 0) {
#pragma unroll
    for (int r = 0; r < 8; ++r) logits[rowBase + half * 8 + r] = part[r] + bo;
  }
}

// ---------------------------------------------------------------------------
// 5) Scatter softmax over edge_ids (2048 segments).
// ---------------------------------------------------------------------------
static __device__ __forceinline__ unsigned fkey(float x) {
  unsigned b = __float_as_uint(x);
  return (b & 0x80000000u) ? ~b : (b | 0x80000000u);   // order-preserving
}
static __device__ __forceinline__ float funkey(unsigned k) {
  unsigned b = (k & 0x80000000u) ? (k & 0x7FFFFFFFu) : ~k;
  return __uint_as_float(b);
}
__global__ void eta_seg_init(unsigned* segkey, float* segsum) {
  const int i = blockIdx.x * blockDim.x + threadIdx.x;
  if (i < kSEG) { segkey[i] = 0u; segsum[i] = 0.f; }
}
__global__ void eta_seg_max(const float* __restrict__ lg, const int* __restrict__ eid,
                            unsigned* __restrict__ segkey) {
  const int i = blockIdx.x * blockDim.x + threadIdx.x;
  if (i < kN) atomicMax(&segkey[eid[i]], fkey(lg[i]));
}
__global__ void eta_seg_sum(const float* __restrict__ lg, const int* __restrict__ eid,
                            const unsigned* __restrict__ segkey,
                            float* __restrict__ ex, float* __restrict__ segsum) {
  const int i = blockIdx.x * blockDim.x + threadIdx.x;
  if (i < kN) {
    const float e = __expf(lg[i] - funkey(segkey[eid[i]]));
    ex[i] = e;
    atomicAdd(&segsum[eid[i]], e);
  }
}
__global__ void eta_seg_norm(const float* __restrict__ ex, const int* __restrict__ eid,
                             const float* __restrict__ segsum, float* __restrict__ out) {
  const int i = blockIdx.x * blockDim.x + threadIdx.x;
  if (i < kN) out[i] = ex[i] / segsum[eid[i]];
}

// ---------------------------------------------------------------------------
extern "C" void kernel_launch(void* const* d_in, const int* in_sizes, int n_in,
                              void* d_out, int out_size, void* d_ws, size_t ws_size,
                              hipStream_t stream) {
  const float* ef   = (const float*)d_in[0];
  const int*   eid  = (const int*)d_in[1];
  const int*   tri  = (const int*)d_in[2];
  const float* Wq   = (const float*)d_in[3];
  const float* bq   = (const float*)d_in[4];
  const float* Wk   = (const float*)d_in[5];
  const float* bk   = (const float*)d_in[6];
  const float* Wv   = (const float*)d_in[7];
  const float* bv   = (const float*)d_in[8];
  const float* W0   = (const float*)d_in[9];
  const float* b0   = (const float*)d_in[10];
  const float* rmsw = (const float*)d_in[11];
  const float* Wres = (const float*)d_in[12];
  const float* bres = (const float*)d_in[13];
  const float* Wout = (const float*)d_in[14];
  const float* bout = (const float*)d_in[15];
  float* out = (float*)d_out;
  (void)in_sizes; (void)n_in; (void)out_size; (void)ws_size;

  char* ws = (char*)d_ws;
  size_t off = 0;
  auto carve = [&](size_t bytes) -> char* {
    char* p = ws + off;
    off += (bytes + 255) & ~(size_t)255;
    return p;
  };
  __bf16*   qbf   = (__bf16*)carve((size_t)kN * kDK * 2);
  __bf16*   kbf   = (__bf16*)carve((size_t)kN * kDK * 2);
  __bf16*   vt    = (__bf16*)carve((size_t)kN * kDK * 2);
  float*    att   = (float*) carve((size_t)kN * kDK * 4);
  float*    Opart = (float*) carve((size_t)kSPLIT * kN * kDK * 4);
  float*    mpart = (float*) carve((size_t)kSPLIT * kN * 4);
  float*    lpart = (float*) carve((size_t)kSPLIT * kN * 4);
  __bf16*   W0t   = (__bf16*)carve((size_t)kDK * kHID * 2);
  __bf16*   Wrt   = (__bf16*)carve((size_t)2 * kHID * kHID * 2);
  float*    lg    = (float*) carve((size_t)kN * 4);
  float*    ex    = (float*) carve((size_t)kN * 4);
  unsigned* sk    = (unsigned*)carve((size_t)kSEG * 4);
  float*    ss    = (float*) carve((size_t)kSEG * 4);

  eta_convw<<<(kDK * kHID + 2 * kHID * kHID + 255) / 256, 256, 0, stream>>>(W0, Wres, W0t, Wrt);
  eta_proj <<<(kN * kDK + 255) / 256, 256, 0, stream>>>(ef, Wq, bq, Wk, bk, Wv, bv, qbf, kbf, vt);
  eta_attn_part<<<dim3(kN / 128, kSPLIT), 256, 0, stream>>>(qbf, kbf, vt, tri, Opart, mpart, lpart);
  eta_attn_merge<<<(kN * kDK + 255) / 256, 256, 0, stream>>>(Opart, mpart, lpart, att);
  eta_mlp  <<<kN / 16, 32, 0, stream>>>(att, W0t, b0, rmsw, Wrt, bres, Wout, bout, lg);
  eta_seg_init<<<(kSEG + 255) / 256, 256, 0, stream>>>(sk, ss);
  eta_seg_max <<<(kN + 255) / 256, 256, 0, stream>>>(lg, eid, sk);
  eta_seg_sum <<<(kN + 255) / 256, 256, 0, stream>>>(lg, eid, sk, ex, ss);
  eta_seg_norm<<<(kN + 255) / 256, 256, 0, stream>>>(ex, eid, ss, out);
}
